// Decoder_59425167507713
// MI455X (gfx1250) — compile-verified, matrix operand order checked by reading, not run on hardware
//
#include <hip/hip_runtime.h>
#include <hip/hip_bf16.h>

// ---------------------------------------------------------------------------
// Transformer decoder (B=32, S=SE=64, D=512, DK=64, L=4, V=8000), fp32 model.
// Roofline: ~56 GFLOP vs ~1.05 GB of out_w traffic -> memory bound on the
// classifier weight stream. fp32 loaded from HBM, converted to f16 in-register,
// f32 accumulation via v_wmma_f32_16x16x32_f16 (CDNA5 wave32 WMMA).
// Each wave computes a 32x16 tile (2 accumulators share one B fragment) so
// out_w is streamed exactly once; classifier B loads are non-temporal.
// ---------------------------------------------------------------------------

#define Bsz 32
#define Sq  64
#define Se  64
#define Dm  512
#define DKh 64
#define Lnum 4
#define Voc 8000
#define BN_SCALE 0.99950037f   // 1/sqrt(1+1e-3)

typedef __attribute__((ext_vector_type(16))) _Float16 v16h;
typedef __attribute__((ext_vector_type(8)))  float    v8f;
typedef __attribute__((ext_vector_type(4)))  float    v4f;

// ---------------------------------------------------------------------------
// Generic wave-level WMMA GEMM: one wave computes a (16*MT)x16 tile of
//   C = alpha * (A @ B) [+ bias] [relu]
// A row-major [M,K]; B row-major [K,N] (TRANSB=0) or row-major [N,K] read as
// B[k][n]=Bt[n][k] (TRANSB=1). Batched via grid.z + strides.
// Fragment layouts per CDNA5 ISA 7.12.2: per-lane K-group kb=(lane>>4)*8,
// elements K=kb+i (i<8) and K=kb+16+i; C/D rows m=(lane>>4)*8+i, col lane&15.
// ---------------------------------------------------------------------------
template<int MT, bool TRANSB, bool RELU, bool BIAS, bool STREAMB>
__global__ __launch_bounds__(32) void gemm_wmma(
    const float* __restrict__ A, const float* __restrict__ Bm,
    const float* __restrict__ bias, float* __restrict__ C,
    int K, int lda, int ldb, int ldc,
    long long sA, long long sB, long long sC, float alpha)
{
    const int lane = threadIdx.x & 31;
    const int n0 = blockIdx.x << 4;
    const int m0 = (blockIdx.y * MT) << 4;
    const int b  = blockIdx.z;
    A  += (long long)b * sA;
    Bm += (long long)b * sB;
    C  += (long long)b * sC;

    const int l15 = lane & 15;            // A row within tile / B,C column
    const int kb  = (lane >> 4) << 3;     // K sub-group base (0 or 8)

    v8f acc[MT];
#pragma unroll
    for (int t = 0; t < MT; ++t) acc[t] = (v8f){};

    for (int k0 = 0; k0 < K; k0 += 32) {
        // ---- B fragment (shared by all MT row-tiles) ----
        v16h fb;
        if (TRANSB) {
            // B[k][n] = Bt[n0+l15][k] : contiguous along K -> b128 loads
            const v4f* p = (const v4f*)(Bm + (size_t)(n0 + l15) * ldb + (k0 + kb));
            v4f b0 = p[0], b1 = p[1], b2 = p[4], b3 = p[5];
#pragma unroll
            for (int i = 0; i < 4; ++i) {
                fb[i]      = (_Float16)b0[i];
                fb[4 + i]  = (_Float16)b1[i];
                fb[8 + i]  = (_Float16)b2[i];
                fb[12 + i] = (_Float16)b3[i];
            }
        } else {
            // B row-major [K,N]: column n0+l15, strided by ldb
            const float* p = Bm + (size_t)(k0 + kb) * ldb + (n0 + l15);
#pragma unroll
            for (int i = 0; i < 8; ++i) {
                float lo = STREAMB ? __builtin_nontemporal_load(p + (size_t)i * ldb)
                                   : p[(size_t)i * ldb];
                float hi = STREAMB ? __builtin_nontemporal_load(p + (size_t)(16 + i) * ldb)
                                   : p[(size_t)(16 + i) * ldb];
                fb[i]     = (_Float16)lo;
                fb[8 + i] = (_Float16)hi;
            }
        }
        // ---- MT A fragments, one WMMA each ----
#pragma unroll
        for (int t = 0; t < MT; ++t) {
            const v4f* p = (const v4f*)(A + (size_t)(m0 + 16 * t + l15) * lda + (k0 + kb));
            v4f a0 = p[0], a1 = p[1], a2 = p[4], a3 = p[5];
            v16h fa;
#pragma unroll
            for (int i = 0; i < 4; ++i) {
                fa[i]      = (_Float16)a0[i];
                fa[4 + i]  = (_Float16)a1[i];
                fa[8 + i]  = (_Float16)a2[i];
                fa[12 + i] = (_Float16)a3[i];
            }
            acc[t] = __builtin_amdgcn_wmma_f32_16x16x32_f16(
                         false, fa, false, fb, (short)0, acc[t], false, false);
        }
    }

    const int mb = (lane >> 4) << 3;      // C row base (0 or 8)
    const float bv = BIAS ? bias[n0 + l15] : 0.0f;
#pragma unroll
    for (int t = 0; t < MT; ++t) {
#pragma unroll
        for (int i = 0; i < 8; ++i) {
            float v = acc[t][i] * alpha + bv;
            if (RELU) v = fmaxf(v, 0.0f);
            C[(size_t)(m0 + 16 * t + mb + i) * ldc + (n0 + l15)] = v;
        }
    }
}

// ---------------------------------------------------------------------------
// Embedding + positional encoding: x[b,s,:] = emb[seq[b,s],:] + pes[s,:]
// ---------------------------------------------------------------------------
__global__ __launch_bounds__(256) void embed_kernel(
    const int* __restrict__ seq, const float* __restrict__ emb,
    const float* __restrict__ pes, float* __restrict__ x)
{
    const int bs = blockIdx.x;          // b*S + s
    const int s  = bs & (Sq - 1);
    const int tok = seq[bs];
    for (int d = threadIdx.x; d < Dm; d += 256)
        x[(size_t)bs * Dm + d] = emb[(size_t)tok * Dm + d] + pes[(size_t)s * Dm + d];
}

// ---------------------------------------------------------------------------
// Row softmax over 64 keys; strict-causal option (keys k < q valid).
// Row q==0 has no valid key -> reference zeroes the head; emit zeros here.
// ---------------------------------------------------------------------------
__global__ __launch_bounds__(64) void softmax64(float* __restrict__ s, int causal)
{
    const int row  = blockIdx.x;        // b*S + q
    const int qpos = row & (Sq - 1);
    const int tid  = threadIdx.x;
    float* r = s + (size_t)row * 64;

    if (causal && qpos == 0) { r[tid] = 0.0f; return; }

    float val = (causal && tid >= qpos) ? -1e9f : r[tid];
    __shared__ float red[64];
    red[tid] = val; __syncthreads();
#pragma unroll
    for (int off = 32; off; off >>= 1) {
        if (tid < off) red[tid] = fmaxf(red[tid], red[tid + off]);
        __syncthreads();
    }
    const float mx = red[0]; __syncthreads();
    const float e = __expf(val - mx);
    red[tid] = e; __syncthreads();
#pragma unroll
    for (int off = 32; off; off >>= 1) {
        if (tid < off) red[tid] += red[tid + off];
        __syncthreads();
    }
    r[tid] = e / red[0];
}

// ---------------------------------------------------------------------------
// Fused residual + inference BatchNorm: x = g*((x + r)*BN_SCALE) + b
// ---------------------------------------------------------------------------
__global__ __launch_bounds__(256) void bn_res(
    float* __restrict__ x, const float* __restrict__ r,
    const float* __restrict__ g, const float* __restrict__ bt, int total)
{
    const int i = blockIdx.x * 256 + threadIdx.x;
    if (i < total) {
        const int d = i & (Dm - 1);
        x[i] = g[d] * ((x[i] + r[i]) * BN_SCALE) + bt[d];
    }
}

// ---------------------------------------------------------------------------
// Host-side dispatch helpers (all waves compute 32x16 tiles: MT=2)
// ---------------------------------------------------------------------------
static inline void gemm(hipStream_t st, const float* A, const float* Bm,
                        const float* bias, float* C, int M, int N, int K,
                        int lda, int ldb, int ldc,
                        long long sA, long long sB, long long sC, int batch,
                        float alpha, bool transb, bool relu, bool streamb)
{
    dim3 grid(N / 16, M / 32, batch), blk(32);
    if (transb)       gemm_wmma<2, true , false, false, false><<<grid, blk, 0, st>>>(A, Bm, bias, C, K, lda, ldb, ldc, sA, sB, sC, alpha);
    else if (relu)    gemm_wmma<2, false, true , true , false><<<grid, blk, 0, st>>>(A, Bm, bias, C, K, lda, ldb, ldc, sA, sB, sC, alpha);
    else if (streamb) gemm_wmma<2, false, false, true , true ><<<grid, blk, 0, st>>>(A, Bm, bias, C, K, lda, ldb, ldc, sA, sB, sC, alpha);
    else if (bias)    gemm_wmma<2, false, false, true , false><<<grid, blk, 0, st>>>(A, Bm, bias, C, K, lda, ldb, ldc, sA, sB, sC, alpha);
    else              gemm_wmma<2, false, false, false, false><<<grid, blk, 0, st>>>(A, Bm, bias, C, K, lda, ldb, ldc, sA, sB, sC, alpha);
}

extern "C" void kernel_launch(void* const* d_in, const int* in_sizes, int n_in,
                              void* d_out, int out_size, void* d_ws, size_t ws_size,
                              hipStream_t stream)
{
    (void)in_sizes; (void)n_in; (void)out_size; (void)ws_size;

    const int*   seq   = (const int*)  d_in[0];
    const float* enc   = (const float*)d_in[1];
    const float* pes   = (const float*)d_in[2];
    const float* emb   = (const float*)d_in[3];
    const float* sa_wq = (const float*)d_in[4];  const float* sa_bq = (const float*)d_in[5];
    const float* sa_wk = (const float*)d_in[6];  const float* sa_bk = (const float*)d_in[7];
    const float* sa_wv = (const float*)d_in[8];  const float* sa_bv = (const float*)d_in[9];
    const float* sa_wo = (const float*)d_in[10]; const float* sa_bo = (const float*)d_in[11];
    const float* ca_wq = (const float*)d_in[12]; const float* ca_bq = (const float*)d_in[13];
    const float* ca_wk = (const float*)d_in[14]; const float* ca_bk = (const float*)d_in[15];
    const float* ca_wv = (const float*)d_in[16]; const float* ca_bv = (const float*)d_in[17];
    const float* ca_wo = (const float*)d_in[18]; const float* ca_bo = (const float*)d_in[19];
    const float* fw1   = (const float*)d_in[20]; const float* fb1   = (const float*)d_in[21];
    const float* fw2   = (const float*)d_in[22]; const float* fb2   = (const float*)d_in[23];
    const float* bn_g  = (const float*)d_in[24]; const float* bn_b  = (const float*)d_in[25];
    const float* out_w = (const float*)d_in[26]; const float* out_b = (const float*)d_in[27];
    float* out = (float*)d_out;

    // Workspace layout (floats): ~28 MB total
    float* ws   = (float*)d_ws;
    float* x    = ws;                       // [B*S, D]      1,048,576
    float* tmp  = x    + Bsz * Sq * Dm;     // [B*S, D]      1,048,576
    float* q    = tmp  + Bsz * Sq * Dm;     // [B*S, DK]       131,072
    float* kbuf = q    + Bsz * Sq * DKh;    // [B*Sk, DK]      131,072
    float* vbuf = kbuf + Bsz * Se * DKh;    // [B*Sk, DK]      131,072
    float* att  = vbuf + Bsz * Se * DKh;    // [B, S, Sk]      131,072
    float* head = att  + Bsz * Sq * Se;     // [B*S, DK]       131,072
    float* ffh  = head + Bsz * Sq * DKh;    // [B*S, 4D]     4,194,304

    const int   MS   = Bsz * Sq;            // 2048 token rows
    const float invs = 0.125f;              // 1/sqrt(DK)

    // x = emb[seq] + pes
    embed_kernel<<<MS, 256, 0, stream>>>(seq, emb, pes, x);

    for (int l = 0; l < Lnum; ++l) {
        const long long oQK = (long long)l * Dm * DKh;   // [D,DK] weight slab
        const long long oO  = (long long)l * DKh * Dm;   // [DK,D] weight slab
        const long long oDK = (long long)l * DKh;
        const long long oD  = (long long)l * Dm;

        // ---------------- self-attention ----------------
        gemm(stream, x, sa_wq + oQK, sa_bq + oDK, q,    MS, DKh, Dm, Dm, DKh, DKh, 0, 0, 0, 1, 1.0f, false, false, false);
        gemm(stream, x, sa_wk + oQK, sa_bk + oDK, kbuf, MS, DKh, Dm, Dm, DKh, DKh, 0, 0, 0, 1, 1.0f, false, false, false);
        gemm(stream, x, sa_wv + oQK, sa_bv + oDK, vbuf, MS, DKh, Dm, Dm, DKh, DKh, 0, 0, 0, 1, 1.0f, false, false, false);
        // scores = (q @ k^T) / sqrt(DK), batched over B
        gemm(stream, q, kbuf, nullptr, att, Sq, Sq, DKh, DKh, DKh, Sq,
             (long long)Sq * DKh, (long long)Sq * DKh, (long long)Sq * Sq, Bsz, invs, true, false, false);
        softmax64<<<MS, 64, 0, stream>>>(att, 1);
        // head = att @ v, batched
        gemm(stream, att, vbuf, nullptr, head, Sq, DKh, Sq, Sq, DKh, DKh,
             (long long)Sq * Sq, (long long)Sq * DKh, (long long)Sq * DKh, Bsz, 1.0f, false, false, false);
        gemm(stream, head, sa_wo + oO, sa_bo + oD, tmp, MS, Dm, DKh, DKh, Dm, Dm, 0, 0, 0, 1, 1.0f, false, false, false);
        bn_res<<<(MS * Dm) / 256, 256, 0, stream>>>(x, tmp, bn_g + (l * 3 + 0) * Dm, bn_b + (l * 3 + 0) * Dm, MS * Dm);

        // ---------------- cross-attention ----------------
        gemm(stream, x,   ca_wq + oQK, ca_bq + oDK, q,    MS, DKh, Dm, Dm, DKh, DKh, 0, 0, 0, 1, 1.0f, false, false, false);
        gemm(stream, enc, ca_wk + oQK, ca_bk + oDK, kbuf, Bsz * Se, DKh, Dm, Dm, DKh, DKh, 0, 0, 0, 1, 1.0f, false, false, false);
        gemm(stream, enc, ca_wv + oQK, ca_bv + oDK, vbuf, Bsz * Se, DKh, Dm, Dm, DKh, DKh, 0, 0, 0, 1, 1.0f, false, false, false);
        gemm(stream, q, kbuf, nullptr, att, Sq, Se, DKh, DKh, DKh, Se,
             (long long)Sq * DKh, (long long)Se * DKh, (long long)Sq * Se, Bsz, invs, true, false, false);
        softmax64<<<MS, 64, 0, stream>>>(att, 0);
        gemm(stream, att, vbuf, nullptr, head, Sq, DKh, Se, Se, DKh, DKh,
             (long long)Sq * Se, (long long)Se * DKh, (long long)Sq * DKh, Bsz, 1.0f, false, false, false);
        gemm(stream, head, ca_wo + oO, ca_bo + oD, tmp, MS, Dm, DKh, DKh, Dm, Dm, 0, 0, 0, 1, 1.0f, false, false, false);
        bn_res<<<(MS * Dm) / 256, 256, 0, stream>>>(x, tmp, bn_g + (l * 3 + 1) * Dm, bn_b + (l * 3 + 1) * Dm, MS * Dm);

        // ---------------- FFN ----------------
        gemm(stream, x, fw1 + (long long)l * Dm * 4 * Dm, fb1 + (long long)l * 4 * Dm, ffh,
             MS, 4 * Dm, Dm, Dm, 4 * Dm, 4 * Dm, 0, 0, 0, 1, 1.0f, false, true, false);   // ReLU fused
        gemm(stream, ffh, fw2 + (long long)l * 4 * Dm * Dm, fb2 + oD, tmp,
             MS, Dm, 4 * Dm, 4 * Dm, Dm, Dm, 0, 0, 0, 1, 1.0f, false, false, false);
        bn_res<<<(MS * Dm) / 256, 256, 0, stream>>>(x, tmp, bn_g + (l * 3 + 2) * Dm, bn_b + (l * 3 + 2) * Dm, MS * Dm);
    }

    // Final classifier: [B, S*D] @ [S*D, V] + out_b.
    // M=32 -> a single 32x16 wave-tile per 16 columns: out_w streamed exactly
    // once from HBM (non-temporal, keeps L2 for activations/weights).
    gemm(stream, x, out_w, out_b, out, Bsz, Voc, Sq * Dm, Sq * Dm, Voc, Voc,
         0, 0, 0, 1, 1.0f, false, false, true);
}
